// MultiHeadAttention_75419625717957
// MI455X (gfx1250) — matrix operand order repaired and checked by reading, not compiled, not run on hardware
//
#include <hip/hip_runtime.h>
#include <stdint.h>

// ---------------------------------------------------------------------------
// MI455X (gfx1250, wave32) implementation.
// Pipeline:
//   K0: f32 -> bf16 conversion of activations (with [L,B,D]->[B*L,D] transpose)
//   K1: f32 -> bf16 conversion of weight matrices
//   K2: QKV projection GEMMs        (v_wmma_f32_16x16x32_bf16)
//   K3: per-(b,h) score GEMM * 1/8  (v_wmma_f32_16x16x32_bf16)
//   K4: softmax over heads (axis=1, the legacy-dim quirk), in place
//   K5: top-100 per row, one wave32 per row, packed-key max reduction
//   K6: gather -> LDS (global_load_async_to_lds_b128, ASYNCcnt) + output GEMM
// ---------------------------------------------------------------------------

typedef __bf16 bf16;
typedef __attribute__((ext_vector_type(16))) __bf16 v16bf;
typedef __attribute__((ext_vector_type(8)))  __bf16 v8bf;
typedef __attribute__((ext_vector_type(8)))  float  v8f;

#define L_DIM   512
#define B_DIM   2
#define D_DIM   512
#define H_DIM   8
#define DT_DIM  64
#define NUM_SEL 100
#define M_PROJ  (B_DIM * L_DIM)               /* 1024   */
#define M_OUT   (B_DIM * L_DIM * NUM_SEL)     /* 102400 */
#define HEAD_ELEMS (L_DIM * DT_DIM)           /* 32768  */
#define ROW_ELEMS  (NUM_SEL * DT_DIM)         /* 6400   */
#define BATCH_G    (L_DIM * NUM_SEL * DT_DIM) /* 3276800 per-head span in raw reshape */
#define LDS_ROW 520                           /* 512 + 8 pad: bank-conflict-free frags */

// ------------------------------ WMMA helpers -------------------------------

__device__ __forceinline__ v8f wmma_bf16(v16bf a, v16bf b, v8f c) {
  // (neg_a, A, neg_b, B, c_mod, C, reuse_a, reuse_b)
  return __builtin_amdgcn_wmma_f32_16x16x32_bf16(false, a, false, b, (short)0, c,
                                                 false, false);
}

// A-matrix 16x32 bf16 fragment: lane%16 = M row; elements 0..7 hold
// K = kb + half*8 + {0..7}, elements 8..15 hold K = kb + 16 + half*8 + {0..7}.
__device__ __forceinline__ v16bf load_fragA(const bf16* rowbase, int kb, int h8) {
  union { v16bf v; v8bf p[2]; } u;
  u.p[0] = *(const v8bf*)(rowbase + kb + h8);
  u.p[1] = *(const v8bf*)(rowbase + kb + 16 + h8);
  return u.v;
}

// B-matrix 32x16 bf16 fragment: lane%16 = N column; elements 0..7 hold
// K = kb + half*8 + {0..7}, elements 8..15 hold K = kb + 16 + half*8 + {0..7}.
// rowbase points at the K-contiguous row for column N.
__device__ __forceinline__ v16bf load_fragB(const bf16* __restrict__ rowbase,
                                            int kb, int lane) {
  int h8 = ((lane >> 4) & 1) << 3;
  union { v16bf v; v8bf p[2]; } u;
  u.p[0] = *(const v8bf*)(rowbase + kb + h8);
  u.p[1] = *(const v8bf*)(rowbase + kb + 16 + h8);
  return u.v;
}

// ------------------------------ K0 / K1: convert ---------------------------

__global__ void cvt_x_kernel(const float* __restrict__ src, bf16* __restrict__ dst) {
  int t = blockIdx.x * 256 + threadIdx.x;   // 0 .. M_PROJ*D_DIM-1
  int m  = t >> 9;
  int kk = t & 511;
  int b  = m >> 9;
  int l  = m & 511;
  dst[t] = (bf16)src[((size_t)l * B_DIM + b) * D_DIM + kk];
}

__global__ void cvt_w_kernel(const float* __restrict__ src, bf16* __restrict__ dst) {
  int t = blockIdx.x * 256 + threadIdx.x;   // 0 .. D_DIM*D_DIM-1
  dst[t] = (bf16)src[t];
}

// --------------------- K2: projection GEMM  Y = X*W^T + b ------------------

__global__ void proj_gemm(const bf16* __restrict__ X, const bf16* __restrict__ W,
                          const float* __restrict__ bias, bf16* __restrict__ Y) {
  int lane = threadIdx.x & 31;
  int tile = blockIdx.x * 8 + (threadIdx.x >> 5);   // 2048 tiles
  int tm = tile >> 5, tn = tile & 31;               // 64 x 32 tile grid
  int ln15 = lane & 15;
  int h8 = ((lane >> 4) & 1) << 3;
  const bf16* arow = X + (size_t)(tm * 16 + ln15) * D_DIM;
  const bf16* brow = W + (size_t)(tn * 16 + ln15) * D_DIM;
  v8f acc = {};
#pragma unroll 4
  for (int kb = 0; kb < D_DIM; kb += 32)
    acc = wmma_bf16(load_fragA(arow, kb, h8), load_fragB(brow, kb, lane), acc);
  float bv = bias[tn * 16 + ln15];
  int mhi = ((lane >> 4) & 1) << 3;
#pragma unroll
  for (int v = 0; v < 8; ++v)
    Y[(size_t)(tm * 16 + mhi + v) * D_DIM + tn * 16 + ln15] = (bf16)(acc[v] + bv);
}

// --------------------- K3: score GEMM  S = (Qh Kh^T)/8 ---------------------

__global__ void score_gemm(const bf16* __restrict__ PQ, const bf16* __restrict__ PK,
                           float* __restrict__ S) {
  int lane = threadIdx.x & 31;
  int tile = blockIdx.x * 8 + (threadIdx.x >> 5);
  int bh = tile >> 10;                     // (b*8+h)
  int t  = tile & 1023;
  int tm = t >> 5, tn = t & 31;
  int ln15 = lane & 15;
  int h8 = ((lane >> 4) & 1) << 3;
  size_t hoff = (size_t)(bh >> 3) * (L_DIM * D_DIM) + (size_t)(bh & 7) * HEAD_ELEMS;
  const bf16* arow = PQ + hoff + (size_t)(tm * 16 + ln15) * DT_DIM;
  const bf16* brow = PK + hoff + (size_t)(tn * 16 + ln15) * DT_DIM;
  v8f acc = {};
  acc = wmma_bf16(load_fragA(arow, 0,  h8), load_fragB(brow, 0,  lane), acc);
  acc = wmma_bf16(load_fragA(arow, 32, h8), load_fragB(brow, 32, lane), acc);
  int mhi = ((lane >> 4) & 1) << 3;
#pragma unroll
  for (int v = 0; v < 8; ++v)
    S[((size_t)bh * L_DIM + tm * 16 + mhi + v) * L_DIM + tn * 16 + ln15] =
        acc[v] * 0.125f;
}

// --------------------- K4: softmax over heads (axis=1), in place -----------

__global__ void softmax_heads(float* __restrict__ S) {
  int t = blockIdx.x * 256 + threadIdx.x;   // 0 .. B*L*L-1
  int b = t >> 18;
  int r = t & (L_DIM * L_DIM - 1);
  float* p = S + (size_t)b * H_DIM * L_DIM * L_DIM + r;
  float v[H_DIM];
  float mx = -3.0e38f;
#pragma unroll
  for (int h = 0; h < H_DIM; ++h) { v[h] = p[(size_t)h * L_DIM * L_DIM]; mx = fmaxf(mx, v[h]); }
  float s = 0.f;
#pragma unroll
  for (int h = 0; h < H_DIM; ++h) { v[h] = __expf(v[h] - mx); s += v[h]; }
  float inv = 1.0f / s;
#pragma unroll
  for (int h = 0; h < H_DIM; ++h) p[(size_t)h * L_DIM * L_DIM] = v[h] * inv;
}

// --------------------- K5: top-100 of each 512-row, one wave32/row ---------

__global__ void topk_kernel(const float* __restrict__ A, int* __restrict__ IDX) {
  int lane = threadIdx.x & 31;
  int row  = blockIdx.x * 8 + (threadIdx.x >> 5);   // 8192 rows
  const float* r = A + (size_t)row * L_DIM;
  float vals[16];
#pragma unroll
  for (int j = 0; j < 16; ++j) vals[j] = r[j * 32 + lane];
  int* out = IDX + (size_t)row * NUM_SEL;
  for (int t = 0; t < NUM_SEL; ++t) {
    unsigned long long key = 0ull;
#pragma unroll
    for (int j = 0; j < 16; ++j) {
      unsigned long long k2 =
          ((unsigned long long)__float_as_uint(vals[j]) << 32) |
          (unsigned)(j * 32 + lane);
      key = (k2 > key) ? k2 : key;
    }
#pragma unroll
    for (int off = 16; off >= 1; off >>= 1) {
      unsigned long long o = __shfl_xor(key, off, 32);
      key = (o > key) ? o : key;
    }
    int gidx = (int)(key & 0x1ffull);
    if (lane == 0) out[(NUM_SEL - 1) - t] = gidx;
    if ((gidx & 31) == lane) {
      int j = gidx >> 5;
#pragma unroll
      for (int jj = 0; jj < 16; ++jj)
        if (jj == j) vals[jj] = 0.0f;     // remove winner (real vals are > 0)
    }
  }
}

// ------------- K6: async gather -> LDS, then output GEMM -------------------
// Block = 64 output rows x all 512 columns (4 M-tiles x 32 N-tiles).
// Stage 1: gather the 64x512 bf16 G-tile into LDS with
//          global_load_async_to_lds_b128 (per-lane scattered 16B chunks),
//          wait ASYNCcnt, barrier.
// Stage 2: each of 8 waves owns 4 N-tiles; per (tn, kb) one B fragment load
//          feeds 4 WMMAs (A fragments re-read cheaply from LDS).
// This reads Wc once per block (0.8 GB total vs 3.3 GB before) and gathers G
// exactly once (105 MB total).

__global__ void out_gemm(const bf16* __restrict__ PV, const bf16* __restrict__ WC,
                         const int* __restrict__ IDX, const float* __restrict__ bc,
                         float* __restrict__ OUT) {
  __shared__ bf16 ltile[64 * LDS_ROW];          // 66,560 B
  const int tid = threadIdx.x;
  const int mBase = blockIdx.x * 64;

  // ---- Stage 1: gather ----
#pragma unroll 4
  for (int i = 0; i < 16; ++i) {
    int chunk = i * 256 + tid;                  // 0..4095 : (row rr, 8-col chunk cc)
    int rr = chunk >> 6;
    int cc = chunk & 63;
    int m  = mBase + rr;
    int b   = m / (L_DIM * NUM_SEL);
    int rem = m - b * (L_DIM * NUM_SEL);
    int lp  = rem / NUM_SEL;
    int np  = rem - lp * NUM_SEL;
    int f   = lp * (NUM_SEL * D_DIM) + np * D_DIM + cc * 8;  // flat reshape index
    int hh  = f / BATCH_G;
    int fr  = f - hh * BATCH_G;
    int l   = fr / ROW_ELEMS;
    int fr2 = fr - l * ROW_ELEMS;
    int n   = fr2 >> 6;
    int dt  = fr2 & 63;
    int srow = IDX[(size_t)b * (H_DIM * L_DIM * NUM_SEL) +
                   (hh * L_DIM + l) * NUM_SEL + n];
    const bf16* src = PV + (size_t)b * (L_DIM * D_DIM) +
                      (size_t)hh * HEAD_ELEMS + srow * DT_DIM + dt;
    unsigned loff = (unsigned)(uintptr_t)(&ltile[rr * LDS_ROW + cc * 8]);
    unsigned long long gaddr = (unsigned long long)(uintptr_t)src;
    asm volatile("global_load_async_to_lds_b128 %0, %1, off"
                 :: "v"(loff), "v"(gaddr) : "memory");
  }
  asm volatile("s_wait_asynccnt 0x0" ::: "memory");
  __syncthreads();

  // ---- Stage 2: GEMM ----
  const int lane = tid & 31;
  const int wave = tid >> 5;
  const int ln15 = lane & 15;
  const int h8  = ((lane >> 4) & 1) << 3;
  const int mhi = ((lane >> 4) & 1) << 3;

#pragma unroll
  for (int tni = 0; tni < 4; ++tni) {
    int tn = wave * 4 + tni;
    const bf16* brow = WC + (size_t)(tn * 16 + ln15) * D_DIM;
    __builtin_prefetch(brow, 0, 3);             // global_prefetch_b8
    v8f acc0 = {}, acc1 = {}, acc2 = {}, acc3 = {};
    for (int kb = 0; kb < D_DIM; kb += 32) {
      v16bf bfrag = load_fragB(brow, kb, lane);
      const bf16* a0 = &ltile[(0 * 16 + ln15) * LDS_ROW];
      const bf16* a1 = &ltile[(1 * 16 + ln15) * LDS_ROW];
      const bf16* a2 = &ltile[(2 * 16 + ln15) * LDS_ROW];
      const bf16* a3 = &ltile[(3 * 16 + ln15) * LDS_ROW];
      acc0 = wmma_bf16(load_fragA(a0, kb, h8), bfrag, acc0);
      acc1 = wmma_bf16(load_fragA(a1, kb, h8), bfrag, acc1);
      acc2 = wmma_bf16(load_fragA(a2, kb, h8), bfrag, acc2);
      acc3 = wmma_bf16(load_fragA(a3, kb, h8), bfrag, acc3);
    }
    float bv = bc[tn * 16 + ln15];
    int col = tn * 16 + ln15;
#pragma unroll
    for (int v = 0; v < 8; ++v) {
      OUT[(size_t)(mBase + 0 * 16 + mhi + v) * D_DIM + col] = acc0[v] + bv;
      OUT[(size_t)(mBase + 1 * 16 + mhi + v) * D_DIM + col] = acc1[v] + bv;
      OUT[(size_t)(mBase + 2 * 16 + mhi + v) * D_DIM + col] = acc2[v] + bv;
      OUT[(size_t)(mBase + 3 * 16 + mhi + v) * D_DIM + col] = acc3[v] + bv;
    }
  }
}

// ------------------------------- launcher ----------------------------------

extern "C" void kernel_launch(void* const* d_in, const int* in_sizes, int n_in,
                              void* d_out, int out_size, void* d_ws, size_t ws_size,
                              hipStream_t stream) {
  const float* q  = (const float*)d_in[0];
  const float* k  = (const float*)d_in[1];
  const float* v  = (const float*)d_in[2];
  const float* Wq = (const float*)d_in[3];
  const float* bq = (const float*)d_in[4];
  const float* Wk = (const float*)d_in[5];
  const float* bk = (const float*)d_in[6];
  const float* Wv = (const float*)d_in[7];
  const float* bv = (const float*)d_in[8];
  const float* Wc = (const float*)d_in[9];
  const float* bc = (const float*)d_in[10];

  char* ws = (char*)d_ws;
  size_t off = 0;
  const size_t XB = (size_t)M_PROJ * D_DIM * sizeof(bf16);   // 1 MB
  const size_t WB = (size_t)D_DIM * D_DIM * sizeof(bf16);    // 512 KB
  bf16* xq = (bf16*)(ws + off); off += XB;
  bf16* xk = (bf16*)(ws + off); off += XB;
  bf16* xv = (bf16*)(ws + off); off += XB;
  bf16* wq = (bf16*)(ws + off); off += WB;
  bf16* wk = (bf16*)(ws + off); off += WB;
  bf16* wv = (bf16*)(ws + off); off += WB;
  bf16* wc = (bf16*)(ws + off); off += WB;
  bf16* pq = (bf16*)(ws + off); off += XB;
  bf16* pk = (bf16*)(ws + off); off += XB;
  bf16* pv = (bf16*)(ws + off); off += XB;
  float* score = (float*)(ws + off);
  off += (size_t)B_DIM * H_DIM * L_DIM * L_DIM * sizeof(float);  // 16 MB
  int* idx = (int*)(ws + off);
  off += (size_t)B_DIM * H_DIM * L_DIM * NUM_SEL * sizeof(int);  // 3.2 MB

  cvt_x_kernel<<<2048, 256, 0, stream>>>(q, xq);
  cvt_x_kernel<<<2048, 256, 0, stream>>>(k, xk);
  cvt_x_kernel<<<2048, 256, 0, stream>>>(v, xv);
  cvt_w_kernel<<<1024, 256, 0, stream>>>(Wq, wq);
  cvt_w_kernel<<<1024, 256, 0, stream>>>(Wk, wk);
  cvt_w_kernel<<<1024, 256, 0, stream>>>(Wv, wv);
  cvt_w_kernel<<<1024, 256, 0, stream>>>(Wc, wc);
  proj_gemm<<<256, 256, 0, stream>>>(xq, wq, bq, pq);
  proj_gemm<<<256, 256, 0, stream>>>(xk, wk, bk, pk);
  proj_gemm<<<256, 256, 0, stream>>>(xv, wv, bv, pv);
  score_gemm<<<2048, 256, 0, stream>>>(pq, pk, score);
  softmax_heads<<<2048, 256, 0, stream>>>(score);
  topk_kernel<<<1024, 256, 0, stream>>>(score, idx);
  out_gemm<<<M_OUT / 64, 256, 0, stream>>>(pv, wc, idx, bc, (float*)d_out);
}
